// PGAU_41394894799299
// MI455X (gfx1250) — compile-verified, matrix operand order checked by reading, not run on hardware
//
#include <hip/hip_runtime.h>
#include <hip/hip_bf16.h>

// ---------------------------------------------------------------------------
// CDNA5 (gfx1250) implementation of the gated-attention-unit reference.
// All GEMMs use v_wmma_f32_16x16x32_f16 (wave32, 16x16 tiles, fp32 accum).
// Staging is branch-free (clamped loads + cndmask selects), vectorized
// (global_load_b128 / ds_store_b64) and software-pipelined one K-tile deep.
// ---------------------------------------------------------------------------

typedef __attribute__((ext_vector_type(16))) _Float16 v16h;
typedef __attribute__((ext_vector_type(8)))  float    v8f;

enum {
    EPI_NONE      = 0,
    EPI_BIAS      = 1,
    EPI_BIAS_SILU = 2,
    EPI_MUL_AUX   = 3,   // C = acc * aux[m][n]      (U * (attn@V))
    EPI_ACCUM     = 4,   // C += acc                 (second half of concat GEMM)
    EPI_ENERGY    = 5    // logits = mask ? (acc + rel)/scale : -9999
};

struct EpiParams {
    const float*         bias;      // [N]
    const float*         aux;       // batched aux matrix (U) for MUL_AUX
    long long            aux_batch; // aux batch stride (elements)
    int                  ld_aux;
    const int*           heights;   // (Nb, S) int32
    const float*         qh;        // (Nb, S0, H) rel-height logits
    const unsigned char* mask;      // (Nb, S, S) bool
    float                inv_scale; // 1/sqrt(2d)
    int                  S, S0, H;
};

constexpr int GLS = 48;   // LDS row stride in halves (96 B: 8/16-B aligned rows)

// ---- branch-free tile loaders: clamped addresses, validity applied later ----

// A tile: 64 rows x 32 k.  vec path requires lda%4==0 && K%32==0.
__device__ __forceinline__ void load_tile_A(const float* __restrict__ A, int lda,
                                            int M, int K, int m0, int k0, int tid,
                                            bool vec, float r[16])
{
    #pragma unroll
    for (int i = 0; i < 4; ++i) {
        const int flat = i * 512 + tid * 4;
        const int rr = flat >> 5, cc = flat & 31;
        int gm = m0 + rr; gm = gm < M ? gm : M - 1;
        const float* __restrict__ row = A + (long long)gm * lda;
        if (vec) {
            const float4 v = *(const float4*)(row + k0 + cc);
            r[i * 4 + 0] = v.x; r[i * 4 + 1] = v.y;
            r[i * 4 + 2] = v.z; r[i * 4 + 3] = v.w;
        } else {
            #pragma unroll
            for (int j = 0; j < 4; ++j) {
                int gk = k0 + cc + j; gk = gk < K ? gk : K - 1;
                r[i * 4 + j] = row[gk];
            }
        }
    }
}

__device__ __forceinline__ void store_tile_A(_Float16* __restrict__ As, int tid,
                                             int M, int K, int m0, int k0,
                                             const float r[16])
{
    #pragma unroll
    for (int i = 0; i < 4; ++i) {
        const int flat = i * 512 + tid * 4;
        const int rr = flat >> 5, cc = flat & 31;
        const bool mok = (m0 + rr) < M;
        union { _Float16 h[4]; unsigned long long u; } pk;
        #pragma unroll
        for (int j = 0; j < 4; ++j) {
            const bool ok = mok && (k0 + cc + j) < K;
            pk.h[j] = (_Float16)(ok ? r[i * 4 + j] : 0.0f);
        }
        *(unsigned long long*)&As[rr * GLS + cc] = pk.u;   // ds_store_b64
    }
}

// B tile, B is (K,N) row-major: vector along n (requires ldb%4==0, N%64==0).
__device__ __forceinline__ void load_tile_Bnn(const float* __restrict__ B, int ldb,
                                              int N, int K, int n0, int k0, int tid,
                                              float r[16])
{
    #pragma unroll
    for (int i = 0; i < 4; ++i) {
        const int flat = i * 512 + tid * 4;        // kk*64 + n, n multiple of 4
        const int kk = flat >> 6, n = flat & 63;
        int gk = k0 + kk; gk = gk < K ? gk : K - 1;
        const float4 v = *(const float4*)(B + (long long)gk * ldb + (n0 + n));
        r[i * 4 + 0] = v.x; r[i * 4 + 1] = v.y;
        r[i * 4 + 2] = v.z; r[i * 4 + 3] = v.w;
    }
}

__device__ __forceinline__ void store_tile_Bnn(_Float16* __restrict__ Bs, int tid,
                                               int N, int K, int n0, int k0,
                                               const float r[16])
{
    #pragma unroll
    for (int i = 0; i < 4; ++i) {
        const int flat = i * 512 + tid * 4;
        const int kk = flat >> 6, n = flat & 63;
        const bool kok = (k0 + kk) < K;
        #pragma unroll
        for (int j = 0; j < 4; ++j) {
            const bool ok = kok && (n0 + n + j) < N;
            Bs[(n + j) * GLS + kk] = (_Float16)(ok ? r[i * 4 + j] : 0.0f);
        }
    }
}

// B tile, B is (N,K) row-major (Q@K^T): vector along k (ldb%4==0, K%32==0).
__device__ __forceinline__ void load_tile_Bt(const float* __restrict__ B, int ldb,
                                             int N, int K, int n0, int k0, int tid,
                                             float r[16])
{
    #pragma unroll
    for (int i = 0; i < 4; ++i) {
        const int flat = i * 512 + tid * 4;        // n*32 + kk, kk multiple of 4
        const int n = flat >> 5, kk = flat & 31;
        int gn = n0 + n; gn = gn < N ? gn : N - 1;
        const float4 v = *(const float4*)(B + (long long)gn * ldb + (k0 + kk));
        r[i * 4 + 0] = v.x; r[i * 4 + 1] = v.y;
        r[i * 4 + 2] = v.z; r[i * 4 + 3] = v.w;
    }
}

__device__ __forceinline__ void store_tile_Bt(_Float16* __restrict__ Bs, int tid,
                                              int N, int K, int n0, int k0,
                                              const float r[16])
{
    #pragma unroll
    for (int i = 0; i < 4; ++i) {
        const int flat = i * 512 + tid * 4;
        const int n = flat >> 5, kk = flat & 31;
        const bool nok = (n0 + n) < N;
        union { _Float16 h[4]; unsigned long long u; } pk;
        #pragma unroll
        for (int j = 0; j < 4; ++j) {
            const bool ok = nok && (k0 + kk + j) < K;
            pk.h[j] = (_Float16)(ok ? r[i * 4 + j] : 0.0f);
        }
        *(unsigned long long*)&Bs[n * GLS + kk] = pk.u;    // ds_store_b64
    }
}

// ---------------------------------------------------------------------------
// Tiled WMMA GEMM:  C[z] = A[z] (M x K) * B[z] (K x N)  + epilogue
// Block = 128 threads = 4 waves; block tile 64x64, K-step 32; one K-tile of
// software pipelining (next tile's global loads issue before the WMMAs).
// ---------------------------------------------------------------------------
template<int EPI, bool BT>
__global__ void __launch_bounds__(128)
gemm_wmma(const float* __restrict__ A, long long Abatch, int lda,
          const float* __restrict__ B, long long Bbatch, int ldb,
          float*       __restrict__ C, long long Cbatch, int ldc,
          int M, int N, int K, EpiParams ep)
{
    constexpr int BM = 64, BK = 32;
    __shared__ _Float16 As[BM * GLS];           // M-major [m][k]
    __shared__ _Float16 Bs[BM * GLS];           // N-major [n][k]

    const int z = blockIdx.z;
    A += (long long)z * Abatch;
    B += (long long)z * Bbatch;
    C += (long long)z * Cbatch;

    const int tid  = threadIdx.x;
    const int lane = tid & 31;
    const int w    = tid >> 5;                  // wave 0..3
    const int wr   = w >> 1, wc = w & 1;        // 2x2 wave grid
    const int m0   = blockIdx.y * BM;
    const int n0   = blockIdx.x * BM;

    const int l15 = lane & 15;
    const int kb  = (lane >> 4) * 8;            // ISA 16-bit A/B K-packing base

    const bool vecA = ((lda & 3) == 0) && ((K & 31) == 0);

    v8f acc[2][2] = {};
    float ra[16], rb[16];

    load_tile_A(A, lda, M, K, m0, 0, tid, vecA, ra);
    if (!BT) load_tile_Bnn(B, ldb, N, K, n0, 0, tid, rb);
    else     load_tile_Bt (B, ldb, N, K, n0, 0, tid, rb);

    for (int k0 = 0; k0 < K; k0 += BK) {
        store_tile_A(As, tid, M, K, m0, k0, ra);
        if (!BT) store_tile_Bnn(Bs, tid, N, K, n0, k0, rb);
        else     store_tile_Bt (Bs, tid, N, K, n0, k0, rb);
        __syncthreads();

        // ---- prefetch / preload next K-tile while WMMAs run ----
        const int kn = k0 + BK;
        if (kn < K) {
            load_tile_A(A, lda, M, K, m0, kn, tid, vecA, ra);
            if (!BT) load_tile_Bnn(B, ldb, N, K, n0, kn, tid, rb);
            else     load_tile_Bt (B, ldb, N, K, n0, kn, tid, rb);
        }
        const int kpf = k0 + 2 * BK;            // L2 prefetch 2 tiles ahead
        if (kpf < K) {
            int pr = m0 + (tid >> 1); pr = pr < M ? pr : M - 1;
            __builtin_prefetch(A + (long long)pr * lda + kpf + (tid & 1) * 16, 0, 1);
        }

        // ---- fragment loads (contiguous 8-half runs -> ds_load_b128) ----
        v16h af[2], bf[2];
        #pragma unroll
        for (int t = 0; t < 2; ++t) {
            const _Float16* ra_ = &As[(wr * 32 + t * 16 + l15) * GLS];
            const _Float16* rb_ = &Bs[(wc * 32 + t * 16 + l15) * GLS];
            #pragma unroll
            for (int e = 0; e < 8; ++e) {
                af[t][e]     = ra_[kb + e];
                af[t][e + 8] = ra_[kb + 16 + e];
                bf[t][e]     = rb_[kb + e];
                bf[t][e + 8] = rb_[kb + 16 + e];
            }
        }
        #pragma unroll
        for (int ti = 0; ti < 2; ++ti)
            #pragma unroll
            for (int tj = 0; tj < 2; ++tj)
                acc[ti][tj] = __builtin_amdgcn_wmma_f32_16x16x32_f16(
                    false, af[ti], false, bf[tj], (short)0, acc[ti][tj],
                    false, false);
        __syncthreads();
    }

    // ---- epilogue + store (C/D layout: VGPR r -> M = r + (lane>>4)*8) ----
    #pragma unroll
    for (int ti = 0; ti < 2; ++ti) {
        #pragma unroll
        for (int tj = 0; tj < 2; ++tj) {
            const int mbase = m0 + wr * 32 + ti * 16 + (lane >> 4) * 8;
            const int n     = n0 + wc * 32 + tj * 16 + l15;
            if (n >= N) continue;
            #pragma unroll
            for (int r = 0; r < 8; ++r) {
                const int m = mbase + r;
                if (m >= M) continue;
                float v = acc[ti][tj][r];
                float* cp = &C[(long long)m * ldc + n];
                if (EPI == EPI_NONE) {
                    *cp = v;
                } else if (EPI == EPI_BIAS) {
                    *cp = v + ep.bias[n];
                } else if (EPI == EPI_BIAS_SILU) {
                    float t2 = v + ep.bias[n];
                    *cp = t2 / (1.0f + __expf(-t2));
                } else if (EPI == EPI_MUL_AUX) {
                    *cp = v * ep.aux[(long long)z * ep.aux_batch +
                                     (long long)m * ep.ld_aux + n];
                } else if (EPI == EPI_ACCUM) {
                    *cp = *cp + v;
                } else if (EPI == EPI_ENERGY) {
                    const int S = ep.S, S0 = ep.S0;
                    int hq  = ep.heights[(long long)z * S + m];
                    int hk  = ep.heights[(long long)z * S + S0 + n];
                    int rel = hk - hq;
                    rel = rel < 1 ? 1 : (rel > ep.H ? ep.H : rel);
                    float re = ep.qh[((long long)z * S0 + m) * ep.H + (rel - 1)];
                    unsigned char mk =
                        ep.mask[(long long)z * S * S + (long long)m * S + (S0 + n)];
                    *cp = mk ? (v + re) * ep.inv_scale : -9999.0f;
                }
            }
        }
    }
}

// ---------------------------------------------------------------------------
// Row LayerNorm, in place.  One 256-thread block per row, D == 256.
// ---------------------------------------------------------------------------
__global__ void __launch_bounds__(256)
ln_kernel(float* __restrict__ x, const float* __restrict__ g,
          const float* __restrict__ b, int D)
{
    __shared__ float red0[8], red1[8];
    const long long row = blockIdx.x;
    const int tid = threadIdx.x;
    float v = x[row * D + tid];
    float s = v, s2 = v * v;
    #pragma unroll
    for (int off = 16; off > 0; off >>= 1) {
        s  += __shfl_down(s,  off, 32);
        s2 += __shfl_down(s2, off, 32);
    }
    if ((tid & 31) == 0) { red0[tid >> 5] = s; red1[tid >> 5] = s2; }
    __syncthreads();
    if (tid == 0) {
        float ts = 0.f, ts2 = 0.f;
        for (int i = 0; i < 8; ++i) { ts += red0[i]; ts2 += red1[i]; }
        red0[0] = ts; red1[0] = ts2;
    }
    __syncthreads();
    const float mean = red0[0] / (float)D;
    const float var  = red1[0] / (float)D - mean * mean;
    const float inv  = rsqrtf(var + 1e-5f);
    x[row * D + tid] = (v - mean) * inv * g[tid] + b[tid];
}

// ---------------------------------------------------------------------------
// Q / K prep: affine-scale Z slices.  grid = Nb*S blocks of d threads.
// ---------------------------------------------------------------------------
__global__ void qk_prep(const float* __restrict__ Z,
                        const float* __restrict__ gam, const float* __restrict__ bet,
                        float* __restrict__ Q, float* __restrict__ Km,
                        int S, int S0, int d)
{
    const int s = blockIdx.x % S;
    const int b = blockIdx.x / S;
    const int k = threadIdx.x;
    const float z = Z[((long long)b * S + s) * d + k];
    if (s < S0)
        Q[((long long)b * S0 + s) * d + k] = z * gam[k] + bet[k];               // row 0
    if (s >= S0)
        Km[((long long)b * (S - S0) + (s - S0)) * d + k] =
            z * gam[2 * d + k] + bet[2 * d + k];                                 // row 2
}

// ---------------------------------------------------------------------------
// Q_h = (Z*g1+b1)[:, :S0] @ embed_heights^T   (tiny: H=10, scalar)
// ---------------------------------------------------------------------------
__global__ void qh_kernel(const float* __restrict__ Z,
                          const float* __restrict__ gam, const float* __restrict__ bet,
                          const float* __restrict__ E, float* __restrict__ Qh,
                          int S, int S0, int d, int H, int total)
{
    const int idx = blockIdx.x * blockDim.x + threadIdx.x;
    if (idx >= total) return;
    const int h = idx % H;
    const int row = idx / H;
    const int b = row / S0, s = row % S0;
    const float* zr = &Z[((long long)b * S + s) * d];
    const float* er = &E[(long long)h * d];
    float acc = 0.f;
    for (int k = 0; k < d; ++k)
        acc += (zr[k] * gam[d + k] + bet[d + k]) * er[k];
    Qh[idx] = acc;
}

// ---------------------------------------------------------------------------
// Row softmax in place, one block per row of length L.
// ---------------------------------------------------------------------------
__global__ void __launch_bounds__(256)
softmax_kernel(float* __restrict__ logits, int L)
{
    __shared__ float red[8];
    float* p = logits + (long long)blockIdx.x * L;
    const int tid = threadIdx.x;

    float mx = -3.0e38f;
    for (int i = tid; i < L; i += 256) mx = fmaxf(mx, p[i]);
    #pragma unroll
    for (int off = 16; off > 0; off >>= 1) mx = fmaxf(mx, __shfl_down(mx, off, 32));
    if ((tid & 31) == 0) red[tid >> 5] = mx;
    __syncthreads();
    if (tid < 8) {
        float v = red[tid];
        #pragma unroll
        for (int off = 4; off > 0; off >>= 1) v = fmaxf(v, __shfl_down(v, off, 8));
        if (tid == 0) red[0] = v;
    }
    __syncthreads();
    mx = red[0];
    __syncthreads();

    float sum = 0.f;
    for (int i = tid; i < L; i += 256) {
        float e = __expf(p[i] - mx);
        p[i] = e;
        sum += e;
    }
    #pragma unroll
    for (int off = 16; off > 0; off >>= 1) sum += __shfl_down(sum, off, 32);
    if ((tid & 31) == 0) red[tid >> 5] = sum;
    __syncthreads();
    if (tid < 8) {
        float v = red[tid];
        #pragma unroll
        for (int off = 4; off > 0; off >>= 1) v += __shfl_down(v, off, 8);
        if (tid == 0) red[0] = v;
    }
    __syncthreads();
    const float inv = 1.0f / red[0];
    for (int i = tid; i < L; i += 256) p[i] *= inv;
}

// ---------------------------------------------------------------------------
// Final gating + residual concat:
//   s < S0: out = sig(gp+bg) * out1 + (1-sig) * seq ;  s >= S0: out = seq
// ---------------------------------------------------------------------------
__global__ void final_kernel(const float* __restrict__ seq,
                             const float* __restrict__ out1,
                             const float* __restrict__ gp,
                             const float* __restrict__ bg,
                             float* __restrict__ out,
                             int Nb, int S, int S0, int D)
{
    const long long idx = (long long)blockIdx.x * blockDim.x + threadIdx.x;
    const long long total = (long long)Nb * S * D;
    if (idx >= total) return;
    const int k = (int)(idx % D);
    const long long rem = idx / D;
    const int s = (int)(rem % S);
    const int b = (int)(rem / S);
    const float r = seq[idx];
    if (s < S0) {
        const long long row = (long long)b * S0 + s;
        const float o = out1[row * D + k];
        const float g = 1.0f / (1.0f + __expf(-(gp[row * D + k] + bg[k])));
        out[idx] = g * o + (1.0f - g) * r;
    } else {
        out[idx] = r;
    }
}

// ---------------------------------------------------------------------------
// Host orchestration
// ---------------------------------------------------------------------------
extern "C" void kernel_launch(void* const* d_in, const int* in_sizes, int n_in,
                              void* d_out, int out_size, void* d_ws, size_t ws_size,
                              hipStream_t stream)
{
    (void)in_sizes; (void)n_in; (void)out_size; (void)ws_size;

    constexpr int Nb = 8, S = 2048, D = 256, d = 128, H = 10;
    constexpr int S0 = (S - 1) / 2;        // 1023
    constexpr int T  = S - S0;             // 1025
    const float inv_scaling = 0.0625f;     // 1/sqrt(2*128)

    const float*         seq     = (const float*)d_in[0];
    const unsigned char* amask   = (const unsigned char*)d_in[1];
    const int*           heights = (const int*)d_in[2];
    const float*         W_init  = (const float*)d_in[3];
    const float*         b_init  = (const float*)d_in[4];
    const float*         ln_g    = (const float*)d_in[5];
    const float*         ln_b    = (const float*)d_in[6];
    const float*         W_U     = (const float*)d_in[7];
    const float*         b_U     = (const float*)d_in[8];
    const float*         W_V     = (const float*)d_in[9];
    const float*         b_V     = (const float*)d_in[10];
    const float*         W_Z     = (const float*)d_in[11];
    const float*         b_Z     = (const float*)d_in[12];
    const float*         os_g    = (const float*)d_in[13];
    const float*         os_b    = (const float*)d_in[14];
    const float*         E_h     = (const float*)d_in[15];
    const float*         W_out   = (const float*)d_in[16];
    const float*         b_out   = (const float*)d_in[17];
    const float*         W_gate  = (const float*)d_in[18];
    const float*         b_gate  = (const float*)d_in[19];
    float*               outp    = (float*)d_out;
    float*               ws      = (float*)d_ws;

    // workspace layout (floats)
    float* xbuf   = ws;                                   // (Nb*S, D)
    float* Zbuf   = xbuf  + (long long)Nb * S  * D;       // (Nb*S, d)
    float* Ubuf   = Zbuf  + (long long)Nb * S  * d;       // (Nb, S0, 2D)
    float* Vbuf   = Ubuf  + (long long)Nb * S0 * 2 * D;   // (Nb, T, 2D)
    float* Qbuf   = Vbuf  + (long long)Nb * T  * 2 * D;   // (Nb, S0, d)
    float* Kbuf   = Qbuf  + (long long)Nb * S0 * d;       // (Nb, T, d)
    float* Qhbuf  = Kbuf  + (long long)Nb * T  * d;       // (Nb, S0, H)
    float* logit  = Qhbuf + (long long)Nb * S0 * H;       // (Nb, S0, T)
    float* UVbuf  = logit + (long long)Nb * S0 * T;       // (Nb, S0, 2D)
    float* out1   = UVbuf + (long long)Nb * S0 * 2 * D;   // (Nb*S0, D)
    float* gp     = out1  + (long long)Nb * S0 * D;       // (Nb*S0, D)

    const dim3 blk(128);
    EpiParams ep0{};  // all-zero defaults

    // 1) xbuf = seq @ W_init + b_init
    {
        EpiParams ep = ep0; ep.bias = b_init;
        gemm_wmma<EPI_BIAS, false><<<dim3(D / 64, (Nb * S) / 64, 1), blk, 0, stream>>>(
            seq, 0, D, W_init, 0, D, xbuf, 0, D, Nb * S, D, D, ep);
    }
    // 2) in-place LayerNorm
    ln_kernel<<<Nb * S, 256, 0, stream>>>(xbuf, ln_g, ln_b, D);

    // 3) U = silu(x[:, :S0] @ W_U + b_U)   (batched over z)
    {
        EpiParams ep = ep0; ep.bias = b_U;
        gemm_wmma<EPI_BIAS_SILU, false><<<dim3((2 * D) / 64, (S0 + 63) / 64, Nb), blk, 0, stream>>>(
            xbuf, (long long)S * D, D, W_U, 0, 2 * D,
            Ubuf, (long long)S0 * 2 * D, 2 * D, S0, 2 * D, D, ep);
    }
    // 4) Z = silu(x @ W_Z + b_Z)
    {
        EpiParams ep = ep0; ep.bias = b_Z;
        gemm_wmma<EPI_BIAS_SILU, false><<<dim3(d / 64, (Nb * S) / 64, 1), blk, 0, stream>>>(
            xbuf, 0, D, W_Z, 0, d, Zbuf, 0, d, Nb * S, d, D, ep);
    }
    // 5) V = silu(x[:, S0:] @ W_V + b_V)
    {
        EpiParams ep = ep0; ep.bias = b_V;
        gemm_wmma<EPI_BIAS_SILU, false><<<dim3((2 * D) / 64, (T + 63) / 64, Nb), blk, 0, stream>>>(
            xbuf + (long long)S0 * D, (long long)S * D, D, W_V, 0, 2 * D,
            Vbuf, (long long)T * 2 * D, 2 * D, T, 2 * D, D, ep);
    }
    // 6) Q / K affine prep
    qk_prep<<<Nb * S, d, 0, stream>>>(Zbuf, os_g, os_b, Qbuf, Kbuf, S, S0, d);

    // 7) Q_h = Qp @ E^T
    {
        const int total = Nb * S0 * H;
        qh_kernel<<<(total + 255) / 256, 256, 0, stream>>>(
            Zbuf, os_g, os_b, E_h, Qhbuf, S, S0, d, H, total);
    }
    // 8) logits = mask ? (Q@K^T + rel)/scale : -9999   (B is N-major -> BT)
    {
        EpiParams ep = ep0;
        ep.heights = heights; ep.qh = Qhbuf; ep.mask = amask;
        ep.inv_scale = inv_scaling; ep.S = S; ep.S0 = S0; ep.H = H;
        gemm_wmma<EPI_ENERGY, true><<<dim3((T + 63) / 64, (S0 + 63) / 64, Nb), blk, 0, stream>>>(
            Qbuf, (long long)S0 * d, d, Kbuf, (long long)T * d, d,
            logit, (long long)S0 * T, T, S0, T, d, ep);
    }
    // 9) softmax in place
    softmax_kernel<<<Nb * S0, 256, 0, stream>>>(logit, T);

    // 10) UV = U * (attn @ V)
    {
        EpiParams ep = ep0;
        ep.aux = Ubuf; ep.aux_batch = (long long)S0 * 2 * D; ep.ld_aux = 2 * D;
        gemm_wmma<EPI_MUL_AUX, false><<<dim3((2 * D) / 64, (S0 + 63) / 64, Nb), blk, 0, stream>>>(
            logit, (long long)S0 * T, T, Vbuf, (long long)T * 2 * D, 2 * D,
            UVbuf, (long long)S0 * 2 * D, 2 * D, S0, 2 * D, T, ep);
    }
    // 11) out1 = UV @ W_out + b_out   (flat over batch)
    {
        EpiParams ep = ep0; ep.bias = b_out;
        gemm_wmma<EPI_BIAS, false><<<dim3(D / 64, (Nb * S0 + 63) / 64, 1), blk, 0, stream>>>(
            UVbuf, 0, 2 * D, W_out, 0, D, out1, 0, D, Nb * S0, D, 2 * D, ep);
    }
    // 12) gp = out1 @ W_gate[:D]  then  gp += res @ W_gate[D:]
    {
        gemm_wmma<EPI_NONE, false><<<dim3(D / 64, (Nb * S0 + 63) / 64, 1), blk, 0, stream>>>(
            out1, 0, D, W_gate, 0, D, gp, 0, D, Nb * S0, D, D, ep0);
        gemm_wmma<EPI_ACCUM, false><<<dim3(D / 64, (S0 + 63) / 64, Nb), blk, 0, stream>>>(
            seq, (long long)S * D, D, W_gate + (long long)D * D, 0, D,
            gp, (long long)S0 * D, D, S0, D, D, ep0);
    }
    // 13) gated blend + residual tail copy into d_out
    {
        const long long total = (long long)Nb * S * D;
        final_kernel<<<(unsigned)((total + 255) / 256), 256, 0, stream>>>(
            seq, out1, gp, b_gate, outp, Nb, S, S0, D);
    }
}
